// TromptCell_16063177687339
// MI455X (gfx1250) — matrix operand order repaired and checked by reading, not compiled, not run on hardware
//
#include <hip/hip_runtime.h>

typedef __attribute__((ext_vector_type(16))) _Float16 v16h;
typedef __attribute__((ext_vector_type(2)))  _Float16 h2;
typedef __attribute__((ext_vector_type(8)))  float    v8f;

#define B_   1024
#define NC_  256     // n_cols
#define NP_  128     // n_prompts
#define D_   256
#define PT_  32      // prompt rows per block

// B-operand pair-packed tile layout in ws: [kblock][d:256][40 halfs]
//   halfs 0..31 = K-pairs (k = kb*32 + kk), halfs 32..39 = pad (bank spread)
#define BT_ROW_H   40
#define BT_TILE_H  (256 * BT_ROW_H)          // 10240 halfs = 20480 B per K-step
#define BT_TILE_B  (BT_TILE_H * 2)

// ---------------------------------------------------------------------------
// CDNA5 async global->LDS copy (ASYNCcnt-tracked), portable via inline asm
// ---------------------------------------------------------------------------
__device__ __forceinline__ void async_b128(unsigned lds_off, const void* g) {
  asm volatile("global_load_async_to_lds_b128 %0, %1, off"
               :: "v"(lds_off), "v"(g) : "memory");
}
__device__ __forceinline__ void async_wait0() {
  asm volatile("s_wait_asynccnt 0x0" ::: "memory");
}

// ---------------------------------------------------------------------------
// block-wide sum reduction (blockDim.x == 256)
// ---------------------------------------------------------------------------
__device__ __forceinline__ float blk_reduce_sum(float v, float* red) {
  int t = threadIdx.x;
  red[t] = v; __syncthreads();
  #pragma unroll
  for (int s = 128; s > 0; s >>= 1) {
    if (t < s) red[t] += red[t + s];
    __syncthreads();
  }
  float r = red[0];
  __syncthreads();
  return r;
}

// ---------------------------------------------------------------------------
// WMMA fragment gathers (wave32, v_wmma_f32_16x16x32_f16 operands in this code's contiguous K order)
// A (16x32 f16): lane = half*16+m ; VGPR v: K = half*16 + 2v+{0,1}
// ---------------------------------------------------------------------------
__device__ __forceinline__ v16h load_a_frag(const _Float16* A, int row0, int k0,
                                            int strideH) {
  int lane = threadIdx.x & 31;
  int hh = lane >> 4, m = lane & 15;
  const _Float16* p0 = A + (row0 + m) * strideH + k0 + hh * 16;
  union { v16h v; unsigned int u[8]; } r;
  const unsigned int* u0 = (const unsigned int*)p0;
  const unsigned int* u1 = (const unsigned int*)(p0 + 8);
  #pragma unroll
  for (int v = 0; v < 4; ++v) { r.u[v] = u0[v]; r.u[4 + v] = u1[v]; }
  return r.v;
}

// B (32x16 f16) from pair-packed column-major LDS tile sB[n][40]:
// lane = half*16+n ; VGPR v: K = half*16 + 2v+{0,1}  (pairs contiguous)
__device__ __forceinline__ v16h load_b_frag(const _Float16* Bt, int col0) {
  int lane = threadIdx.x & 31;
  int hh = lane >> 4, n = lane & 15;
  const unsigned int* u0 =
      (const unsigned int*)(Bt + (col0 + n) * BT_ROW_H + hh * 16);
  union { v16h v; unsigned int u[8]; } r;
  #pragma unroll
  for (int v = 0; v < 8; ++v) r.u[v] = u0[v];
  return r.v;
}

// ---------------------------------------------------------------------------
// prep kernels
// ---------------------------------------------------------------------------
// LN(emb_column)^T, written pair-packed:  colp[(d>>5)*256 + n][d&31]
__global__ __launch_bounds__(256) void prep_col(const float* __restrict__ ec,
                                                const float* __restrict__ g,
                                                const float* __restrict__ bb,
                                                _Float16* __restrict__ colp) {
  __shared__ float red[256];
  int n = blockIdx.x, d = threadIdx.x;
  float v = ec[n * D_ + d];
  float mean = blk_reduce_sum(v, red) * (1.f / 256.f);
  float dv = v - mean;
  float var = blk_reduce_sum(dv * dv, red) * (1.f / 256.f);
  float rstd = rsqrtf(var + 1e-5f);
  float cv = dv * rstd * g[d] + bb[d];
  colp[(size_t)((d >> 5) * 256 + n) * BT_ROW_H + (d & 31)] = (_Float16)cv;
}

// base = LN(emb_prompt) @ W1^T + dense_imp_b + emb_prompt   (batch-invariant)
__global__ __launch_bounds__(256) void prep_base(const float* __restrict__ ep,
                                                 const float* __restrict__ g,
                                                 const float* __restrict__ bb,
                                                 const float* __restrict__ W,
                                                 const float* __restrict__ dib,
                                                 float* __restrict__ base) {
  __shared__ float red[256];
  __shared__ float lnp[256];
  int p = blockIdx.x, d = threadIdx.x;
  float v = ep[p * D_ + d];
  float mean = blk_reduce_sum(v, red) * (1.f / 256.f);
  float dv = v - mean;
  float var = blk_reduce_sum(dv * dv, red) * (1.f / 256.f);
  float rstd = rsqrtf(var + 1e-5f);
  lnp[d] = dv * rstd * g[d] + bb[d];
  __syncthreads();
  const float* wr = W + (size_t)d * (2 * D_);   // W1 half: k < 256
  float s = dib[d];
  #pragma unroll 8
  for (int k = 0; k < D_; ++k) s += lnp[k] * wr[k];
  base[p * D_ + d] = s + v;
}

// W2 (= dense_imp_w[:,256:512]) transposed + pair-packed:
//   w2p[(k>>5)*256 + d][k&31] = W[d][256+k]
__global__ __launch_bounds__(256) void prep_w2(const float* __restrict__ W,
                                               _Float16* __restrict__ w2p) {
  int d = blockIdx.x, k = threadIdx.x;           // coalesced read of W row
  float v = W[(size_t)d * (2 * D_) + D_ + k];
  w2p[(size_t)((k >> 5) * 256 + d) * BT_ROW_H + (k & 31)] = (_Float16)v;
}

// per-(b,n) LayerNorm mean / rstd of x_emb rows
__global__ __launch_bounds__(256) void prep_mr(const float* __restrict__ x,
                                               const float* __restrict__ few,
                                               const float* __restrict__ feb,
                                               float* __restrict__ mr) {
  __shared__ float red[256];
  int n = blockIdx.x, b = blockIdx.y, d = threadIdx.x;
  float xv = x[(size_t)b * NC_ + n];
  float e = xv * few[n * D_ + d] + feb[n * D_ + d];
  e = e > 0.f ? e : 0.f;
  float mean = blk_reduce_sum(e, red) * (1.f / 256.f);
  float dv = e - mean;
  float var = blk_reduce_sum(dv * dv, red) * (1.f / 256.f);
  if (d == 0) {
    mr[((size_t)b * NC_ + n) * 2 + 0] = mean;
    mr[((size_t)b * NC_ + n) * 2 + 1] = rsqrtf(var + 1e-5f);
  }
}

// ---------------------------------------------------------------------------
// fused main kernel: one (batch, 32-prompt tile) per block, 8 waves.
// GEMM1 (xp) -> GEMM2 (logits) -> softmax -> GEMM3 (V), LDS-resident chain.
// B tiles arrive via global_load_async_to_lds_b128 from pre-swizzled ws.
// ---------------------------------------------------------------------------
__global__ __launch_bounds__(256) void trompt_main(
    const float* __restrict__ xg, const float* __restrict__ prev,
    const float* __restrict__ few, const float* __restrict__ feb,
    const float* __restrict__ lng, const float* __restrict__ lnb,
    const _Float16* __restrict__ colp, const _Float16* __restrict__ w2p,
    const float* __restrict__ baseP, const float* __restrict__ mr,
    const float* __restrict__ ew, const float* __restrict__ eb,
    float* __restrict__ out) {
  __shared__ _Float16 bufA[PT_ * 264];   // A operand (xp, then mask), 16.9 KB
  __shared__ float    bufF[PT_ * 264];   // logits f32; overlaid with sB/sA1
  __shared__ float    rowred[PT_ * 2];

  _Float16* sB  = (_Float16*)bufF;               // [256][40] pair-packed B tile
  _Float16* sA1 = (_Float16*)bufF + BT_TILE_H;   // [32][40] GEMM1 A staging
  const unsigned sB_off = (unsigned)(size_t)(void*)sB;

  const int tid  = threadIdx.x;
  const int lane = tid & 31;
  const int wv   = tid >> 5;
  const int wm   = wv >> 2;                      // 0..1 : 16-row group
  const int wn   = wv & 3;                       // 0..3 : 64-col group
  const int b    = blockIdx.y;
  const int p0   = blockIdx.x * PT_;

  v8f acc[4];

  // ---------------- GEMM1: xp = prev @ W2^T (+ base) ----------------
  #pragma unroll
  for (int j = 0; j < 4; ++j)
    #pragma unroll
    for (int v = 0; v < 8; ++v) acc[j][v] = 0.f;

  const float* pc = prev + ((size_t)b * NP_ + p0) * D_;
  for (int k0 = 0; k0 < D_; k0 += 32) {
    const int kb = k0 >> 5;
    __syncthreads();
    {   // stage A 32x32 (f32 global -> f16 LDS, b64 stores)
      int r = tid >> 3, c = (tid & 7) * 4;
      const float4 f = *(const float4*)(pc + r * D_ + k0 + c);
      _Float16* d = sA1 + r * BT_ROW_H + c;
      d[0] = (_Float16)f.x; d[1] = (_Float16)f.y;
      d[2] = (_Float16)f.z; d[3] = (_Float16)f.w;
    }
    {   // async-copy pre-swizzled W2 tile into sB (80 B / thread)
      const char* g = (const char*)w2p + (size_t)kb * BT_TILE_B + tid * 80;
      unsigned l = sB_off + tid * 80;
      #pragma unroll
      for (int i = 0; i < 5; ++i) async_b128(l + i * 16, g + i * 16);
      async_wait0();
    }
    __syncthreads();
    v16h a = load_a_frag(sA1, wm * 16, 0, BT_ROW_H);
    #pragma unroll
    for (int j = 0; j < 4; ++j) {
      v16h bf = load_b_frag(sB, wn * 64 + j * 16);
      acc[j] = __builtin_amdgcn_wmma_f32_16x16x32_f16(false, a, false, bf,
                                                      (short)0, acc[j], false, false);
    }
  }

  __syncthreads();
  {   // xp (+base) -> bufA f16
    int n = lane & 15, hh = lane >> 4;
    #pragma unroll
    for (int v = 0; v < 8; ++v) {
      int row = wm * 16 + hh * 8 + v;
      const float* bp = baseP + (size_t)(p0 + row) * D_;
      #pragma unroll
      for (int j = 0; j < 4; ++j) {
        int col = wn * 64 + j * 16 + n;
        bufA[row * 264 + col] = (_Float16)(acc[j][v] + bp[col]);
      }
    }
  }

  // ---------------- GEMM2: logits = xp @ LN(col)^T ----------------
  #pragma unroll
  for (int j = 0; j < 4; ++j)
    #pragma unroll
    for (int v = 0; v < 8; ++v) acc[j][v] = 0.f;

  for (int k0 = 0; k0 < D_; k0 += 32) {
    const int kb = k0 >> 5;
    __syncthreads();
    {
      const char* g = (const char*)colp + (size_t)kb * BT_TILE_B + tid * 80;
      unsigned l = sB_off + tid * 80;
      #pragma unroll
      for (int i = 0; i < 5; ++i) async_b128(l + i * 16, g + i * 16);
      async_wait0();
    }
    __syncthreads();
    v16h a = load_a_frag(bufA, wm * 16, k0, 264);
    #pragma unroll
    for (int j = 0; j < 4; ++j) {
      v16h bf = load_b_frag(sB, wn * 64 + j * 16);
      acc[j] = __builtin_amdgcn_wmma_f32_16x16x32_f16(false, a, false, bf,
                                                      (short)0, acc[j], false, false);
    }
  }

  __syncthreads();              // sB reads done -> safe to overwrite with logits
  {
    int n = lane & 15, hh = lane >> 4;
    #pragma unroll
    for (int v = 0; v < 8; ++v) {
      int row = wm * 16 + hh * 8 + v;
      #pragma unroll
      for (int j = 0; j < 4; ++j)
        bufF[row * 264 + wn * 64 + j * 16 + n] = acc[j][v];
    }
  }
  __syncthreads();

  // ---------------- softmax over N=256, mask -> bufA f16 ----------------
  if (tid < PT_) {
    const float* rp = bufF + tid * 264;
    float mx = -3.4e38f;
    for (int c = 0; c < NC_; ++c) mx = fmaxf(mx, rp[c]);
    float s = 0.f;
    for (int c = 0; c < NC_; ++c) s += __expf(rp[c] - mx);
    rowred[tid * 2 + 0] = mx;
    rowred[tid * 2 + 1] = 1.f / s;
  }
  __syncthreads();
  {
    int row = tid >> 3, c0 = (tid & 7) * 32;
    float mx = rowred[row * 2], rs = rowred[row * 2 + 1];
    const float* rp = bufF + row * 264 + c0;
    _Float16* dst = bufA + row * 264 + c0;
    #pragma unroll
    for (int i = 0; i < 32; i += 2) {
      h2 hv;
      hv[0] = (_Float16)(__expf(rp[i]     - mx) * rs);
      hv[1] = (_Float16)(__expf(rp[i + 1] - mx) * rs);
      *(h2*)(dst + i) = hv;
    }
  }

  // ---------------- GEMM3: V = mask @ x_emb ----------------
  #pragma unroll
  for (int j = 0; j < 4; ++j)
    #pragma unroll
    for (int v = 0; v < 8; ++v) acc[j][v] = 0.f;

  for (int n0 = 0; n0 < NC_; n0 += 32) {
    __syncthreads();            // orders mask writes / bufF reads before clobber
    {   // build x_emb rows n0..n0+31 on the fly, pair-packed b32 stores
      int kk2 = (tid >> 4) * 2;             // even K pair
      int d0  = (tid & 15) * 16;
      int n   = n0 + kk2;
      float xv0 = xg[(size_t)b * NC_ + n];
      float xv1 = xg[(size_t)b * NC_ + n + 1];
      float m0 = mr[((size_t)b * NC_ + n) * 2 + 0];
      float r0 = mr[((size_t)b * NC_ + n) * 2 + 1];
      float m1 = mr[((size_t)b * NC_ + n + 1) * 2 + 0];
      float r1 = mr[((size_t)b * NC_ + n + 1) * 2 + 1];
      const float* w0 = few + (size_t)n * D_ + d0;
      const float* w1 = w0 + D_;
      const float* c0 = feb + (size_t)n * D_ + d0;
      const float* c1 = c0 + D_;
      #pragma unroll
      for (int i = 0; i < 16; ++i) {
        int d = d0 + i;
        float gd = lng[d], bd = lnb[d];
        float e0 = fmaxf(xv0 * w0[i] + c0[i], 0.f);
        float e1 = fmaxf(xv1 * w1[i] + c1[i], 0.f);
        h2 hv;
        hv[0] = (_Float16)((e0 - m0) * r0 * gd + bd);
        hv[1] = (_Float16)((e1 - m1) * r1 * gd + bd);
        *(h2*)(sB + d * BT_ROW_H + kk2) = hv;
      }
    }
    __syncthreads();
    v16h a = load_a_frag(bufA, wm * 16, n0, 264);
    #pragma unroll
    for (int j = 0; j < 4; ++j) {
      v16h bf = load_b_frag(sB, wn * 64 + j * 16);
      acc[j] = __builtin_amdgcn_wmma_f32_16x16x32_f16(false, a, false, bf,
                                                      (short)0, acc[j], false, false);
    }
  }

  // ---------------- epilogue: V * (1+ew[p]) + eb[p] ----------------
  {
    int n = lane & 15, hh = lane >> 4;
    #pragma unroll
    for (int v = 0; v < 8; ++v) {
      int row = wm * 16 + hh * 8 + v;
      int p = p0 + row;
      float sc = 1.f + ew[p];
      float sh = eb[p];
      float* op = out + ((size_t)b * NP_ + p) * D_;
      #pragma unroll
      for (int j = 0; j < 4; ++j) {
        int col = wn * 64 + j * 16 + n;
        op[col] = acc[j][v] * sc + sh;
      }
    }
  }
}

// ---------------------------------------------------------------------------
extern "C" void kernel_launch(void* const* d_in, const int* in_sizes, int n_in,
                              void* d_out, int out_size, void* d_ws, size_t ws_size,
                              hipStream_t stream) {
  (void)in_sizes; (void)n_in; (void)out_size; (void)ws_size;
  const float* x    = (const float*)d_in[0];
  const float* prev = (const float*)d_in[1];
  const float* few  = (const float*)d_in[2];
  const float* feb  = (const float*)d_in[3];
  const float* lng  = (const float*)d_in[4];
  const float* lnb  = (const float*)d_in[5];
  const float* lcg  = (const float*)d_in[6];
  const float* lcb  = (const float*)d_in[7];
  const float* lpg  = (const float*)d_in[8];
  const float* lpb  = (const float*)d_in[9];
  const float* diw  = (const float*)d_in[10];
  const float* dib  = (const float*)d_in[11];
  const float* ecol = (const float*)d_in[12];
  const float* eprm = (const float*)d_in[13];
  const float* ew   = (const float*)d_in[14];
  const float* eb   = (const float*)d_in[15];
  float* out = (float*)d_out;

  // workspace: w2p 160KB | colp 160KB | base 128KB | mr 2MB   (~2.44 MB)
  char* ws = (char*)d_ws;
  _Float16* w2p  = (_Float16*)(ws);
  _Float16* colp = (_Float16*)(ws + 163840);
  float*    base = (float*)   (ws + 327680);
  float*    mr   = (float*)   (ws + 458752);

  prep_col <<<NC_, 256, 0, stream>>>(ecol, lcg, lcb, colp);
  prep_base<<<NP_, 256, 0, stream>>>(eprm, lpg, lpb, diw, dib, base);
  prep_w2  <<<D_,  256, 0, stream>>>(diw, w2p);
  prep_mr  <<<dim3(NC_, B_), 256, 0, stream>>>(x, few, feb, mr);

  dim3 grid(NP_ / PT_, B_);                     // 4 x 1024 blocks
  trompt_main<<<grid, 256, 0, stream>>>(x, prev, few, feb, lng, lnb,
                                        colp, w2p, base, mr, ew, eb, out);
}
